// CVRS_3435973837463
// MI455X (gfx1250) — matrix operand order repaired and checked by reading, not compiled, not run on hardware
//
#include <hip/hip_runtime.h>
#include <hip/hip_bf16.h>

// Shapes from the reference
#define Bsz 256
#define Lct 3
#define Cct 8
#define Vct 20
#define Sct 32
#define Dct 64

typedef __attribute__((ext_vector_type(16))) _Float16 v16h;
typedef __attribute__((ext_vector_type(8)))  float    v8f;

__device__ __forceinline__ float sigm(float x) { return 1.f / (1.f + expf(-x)); }

// ---- K0: WMMA projection: OUT(B,128) = X(B,64) @ Wg(128,64)^T ----------------
// One wave per 16 rows of X; all 128 output columns.
// A-fragment (16x32 f16): M = lane&15; lane-half selects K half
//   (K = kh*16 + e, contiguous).
// B-fragment (32x16 f16): lanes 0-15 hold K=0..15, lanes 16-31 K=16..31,
//   N = lane&15.  Since B = Wg^T, element e is Wg[N][kbase+e]: contiguous.
__global__ __launch_bounds__(32) void CVRS_proj(
    const float* __restrict__ X, const float* __restrict__ Wg,
    float* __restrict__ OUT) {
  const int lane = threadIdx.x;
  const int M0   = blockIdx.x * 16;
  const int mrow = lane & 15;
  const int kh   = lane >> 4;

  v16h a[2];
#pragma unroll
  for (int kblk = 0; kblk < 2; ++kblk) {
#pragma unroll
    for (int e = 0; e < 16; ++e) {
      const int K = kblk * 32 + kh * 16 + e;
      a[kblk][e] = (_Float16)X[(M0 + mrow) * Dct + K];
    }
  }
#pragma unroll
  for (int n = 0; n < 8; ++n) {
    const int N = n * 16 + mrow;
    v8f acc = {};
#pragma unroll
    for (int kblk = 0; kblk < 2; ++kblk) {
      const int kbase = kblk * 32 + (kh << 4);
      v16h bv;
#pragma unroll
      for (int e = 0; e < 16; ++e) bv[e] = (_Float16)Wg[N * Dct + kbase + e];
      acc = __builtin_amdgcn_wmma_f32_16x16x32_f16(
          false, a[kblk], false, bv, (short)0, acc, false, false);
    }
    const int row = M0 + kh * 8;   // C/D: M = 8*(lane>=16)+i, N = lane&15
#pragma unroll
    for (int i = 0; i < 8; ++i) OUT[(long)(row + i) * 128 + N] = acc[i];
  }
}

// ---- K1: user context: vc, cce_user, vce -------------------------------------
__global__ __launch_bounds__(64) void CVRS_user_ctx(
    const int* __restrict__ ucv, const int* __restrict__ ucl,
    const float* __restrict__ E, const float* __restrict__ Wc,
    const float* __restrict__ Wv, float* __restrict__ cce_u,
    float* __restrict__ vce_o) {
  __shared__ float vc[Cct][Dct];
  __shared__ float vcm[Dct];
  const int b = blockIdx.x, d = threadIdx.x;
#pragma unroll
  for (int c = 0; c < Cct; ++c) {
    float a = 0.f;
    for (int v = 0; v < Vct; ++v)
      a += E[(long)ucv[(b * Cct + c) * Vct + v] * Dct + d];
    vc[c][d] = a;
  }
  __syncthreads();
  float cceacc = 0.f, vcmv = 0.f;
#pragma unroll
  for (int c = 0; c < Cct; ++c) {
    float s = 0.f;
    for (int k = 0; k < Dct; ++k) s += vc[c][k] * Wc[k * Dct + d];
    cceacc += E[(long)ucl[b * Cct + c] * Dct + d] + sigm(s);
    vcmv += vc[c][d];
  }
  cce_u[b * Dct + d] = cceacc * (1.f / Cct);
  vcm[d] = vcmv * (1.f / Cct);
  __syncthreads();
  float s2 = 0.f;
  for (int k = 0; k < Dct; ++k) s2 += vcm[k] * Wv[k * Dct + d];
  vce_o[b * Dct + d] = sigm(s2);
}

// ---- K2: item context: cce = ie * (1 + mean(E[icl[:,1:]])) -------------------
__global__ __launch_bounds__(64) void CVRS_item_ctx(
    const int* __restrict__ items, const int* __restrict__ icl,
    const float* __restrict__ E, float* __restrict__ cce_o) {
  const int b = blockIdx.x, d = threadIdx.x;
  const float ie = E[(long)items[b] * Dct + d];
  float m = 0.f;
#pragma unroll
  for (int c = 1; c < Cct; ++c) m += E[(long)icl[b * Cct + c] * Dct + d];
  m *= (1.f / (Cct - 1));
  cce_o[b * Dct + d] = ie * (1.f + m);
}

// ---- K3: layer_emb. One wave per (b,l,c) tile. Gather-bound. -----------------
// score1[s] = [h|r]·wcc[b], score2[s] = [h|t]·wvv[b] (user only).
__global__ __launch_bounds__(32) void CVRS_layer_emb(
    const int* __restrict__ ts, const float* __restrict__ E,
    const float* __restrict__ R, const float* __restrict__ wcc,
    const float* __restrict__ wvv, float* __restrict__ o_out, int useHT) {
  __shared__ float wc[128];
  __shared__ float wv[128];
  __shared__ float T[Sct][Dct + 1];   // +1: conflict-free column reads
  __shared__ float ebuf[Sct];
  __shared__ float wbuf[Sct];

  const int lane = threadIdx.x;
  const int blk  = blockIdx.x;             // (b*L + l)*C + c
  const int b    = blk / (Lct * Cct);

  const long tbase = (long)blk * 3 * Sct;  // (B,L,C,3,S) layout
  const int hid  = ts[tbase + lane];
  const int rid  = ts[tbase + Sct + lane];
  const int tid2 = ts[tbase + 2 * Sct + lane];

  ((float4*)wc)[lane] = ((const float4*)(wcc + (long)b * 128))[lane];
  if (useHT)
    ((float4*)wv)[lane] = ((const float4*)(wvv + (long)b * 128))[lane];
  __syncthreads();

  const float4* h4 = (const float4*)(E + (long)hid * Dct);
  const float4* r4 = (const float4*)(R + (long)rid * Dct);
  const float4* t4 = (const float4*)(E + (long)tid2 * Dct);
  float s1 = 0.f, s2 = 0.f;
#pragma unroll
  for (int g = 0; g < 16; ++g) {
    const float4 hv = h4[g];
    const float4 rv = r4[g];
    const float4 tv = t4[g];
    s1 += hv.x * wc[4 * g] + hv.y * wc[4 * g + 1] +
          hv.z * wc[4 * g + 2] + hv.w * wc[4 * g + 3];
    s1 += rv.x * wc[64 + 4 * g] + rv.y * wc[64 + 4 * g + 1] +
          rv.z * wc[64 + 4 * g + 2] + rv.w * wc[64 + 4 * g + 3];
    if (useHT) {
      s2 += hv.x * wv[4 * g] + hv.y * wv[4 * g + 1] +
            hv.z * wv[4 * g + 2] + hv.w * wv[4 * g + 3];
      s2 += tv.x * wv[64 + 4 * g] + tv.y * wv[64 + 4 * g + 1] +
            tv.z * wv[64 + 4 * g + 2] + tv.w * wv[64 + 4 * g + 3];
    }
    T[lane][4 * g]     = tv.x;
    T[lane][4 * g + 1] = tv.y;
    T[lane][4 * g + 2] = tv.z;
    T[lane][4 * g + 3] = tv.w;
  }
  ebuf[lane] = expf(sigm(s1));             // softmax(sigmoid(.)) numerator
  __syncthreads();
  float sume = 0.f;
#pragma unroll 8
  for (int s = 0; s < Sct; ++s) sume += ebuf[s];
  float w = ebuf[lane] / sume;
  if (useHT) w *= sigm(s2);
  wbuf[lane] = w;
  __syncthreads();

#pragma unroll
  for (int d = lane; d < Dct; d += 32) {
    float acc = 0.f;
#pragma unroll 8
    for (int s = 0; s < Sct; ++s) acc += wbuf[s] * T[s][d];
    o_out[(long)blk * Dct + d] = acc;
  }
}

// ---- K4: user finalize: mean_c sum_l normalize(o[b,l,c,:]) -------------------
__global__ __launch_bounds__(64) void CVRS_user_final(
    const float* __restrict__ o_u, float* __restrict__ u_fin) {
  __shared__ float red[Dct];
  const int b = blockIdx.x, d = threadIdx.x;
  float acc = 0.f;
  for (int lc = 0; lc < Lct * Cct; ++lc) {
    const float v = o_u[((long)b * Lct * Cct + lc) * Dct + d];
    red[d] = v * v;
    __syncthreads();
    float s = 0.f;
    for (int k = 0; k < Dct; ++k) s += red[k];
    __syncthreads();
    acc += v / fmaxf(sqrtf(s), 1e-12f);
  }
  u_fin[b * Dct + d] = acc * (1.f / Cct);
}

// ---- K5: item finalize + score -----------------------------------------------
__global__ __launch_bounds__(64) void CVRS_item_final(
    const float* __restrict__ o_i, const int* __restrict__ items,
    const float* __restrict__ E, const float* __restrict__ u_fin,
    float* __restrict__ out) {
  __shared__ float cv[Cct][Dct];
  __shared__ float red[Dct];
  __shared__ float pcw[Cct];
  const int b = blockIdx.x, d = threadIdx.x;
  for (int c = 0; c < Cct; ++c) {
    float a = 0.f;
    for (int l = 0; l < Lct; ++l) {
      const float v = o_i[(((long)b * Lct + l) * Cct + c) * Dct + d];
      red[d] = v * v;
      __syncthreads();
      float s = 0.f;
      for (int k = 0; k < Dct; ++k) s += red[k];
      __syncthreads();
      a += v / fmaxf(sqrtf(s), 1e-12f);
    }
    cv[c][d] = a;
  }
  __syncthreads();
  if (d < Cct - 1) {
    float dot = 0.f;
    for (int k = 0; k < Dct; ++k) dot += cv[1 + d][k] * cv[0][k];
    pcw[d] = sigm(dot);
  }
  __syncthreads();
  float mx = pcw[0];
  for (int c = 1; c < Cct - 1; ++c) mx = fmaxf(mx, pcw[c]);
  float sme = 0.f;
  for (int c = 0; c < Cct - 1; ++c) sme += expf(pcw[c] - mx);
  float mcf = 0.f;
  for (int c = 0; c < Cct - 1; ++c)
    mcf += (expf(pcw[c] - mx) / sme) * cv[1 + c][d];
  const float ie = E[(long)items[b] * Dct + d];
  red[d] = ie * mcf;
  __syncthreads();
  float dp = 0.f;
  for (int k = 0; k < Dct; ++k) dp += red[k];
  __syncthreads();
  const float alpha = sigm(dp);
  const float fin = alpha / (1.f + alpha) * mcf + cv[0][d] + ie;
  red[d] = u_fin[b * Dct + d] * fin;
  __syncthreads();
  float sc = 0.f;
  for (int k = 0; k < Dct; ++k) sc += red[k];
  if (d == 0) out[b] = sc;
}

extern "C" void kernel_launch(void* const* d_in, const int* in_sizes, int n_in,
                              void* d_out, int out_size, void* d_ws, size_t ws_size,
                              hipStream_t stream) {
  const int*   ucv   = (const int*)d_in[0];
  const int*   items = (const int*)d_in[1];
  const int*   nitem = (const int*)d_in[2];
  const int*   uts   = (const int*)d_in[3];
  const int*   its   = (const int*)d_in[4];
  const int*   nts   = (const int*)d_in[5];
  const int*   ucl   = (const int*)d_in[6];
  const int*   icl   = (const int*)d_in[7];
  const int*   nicl  = (const int*)d_in[8];
  const float* E     = (const float*)d_in[9];
  const float* R     = (const float*)d_in[10];
  const float* W     = (const float*)d_in[11];
  const float* Wht   = (const float*)d_in[12];
  const float* Wc    = (const float*)d_in[13];
  const float* Wv    = (const float*)d_in[14];
  float* out = (float*)d_out;

  const int B = in_sizes[1];               // 256
  float* ws = (float*)d_ws;
  const size_t vecN  = (size_t)B * Dct;
  const size_t projN = (size_t)B * 128;
  const size_t oN    = (size_t)B * Lct * Cct * Dct;
  float* cce_u  = ws;
  float* vceB   = ws + vecN;
  float* cce_ip = ws + 2 * vecN;
  float* cce_in = ws + 3 * vecN;
  float* wcc_u  = ws + 4 * vecN;
  float* wvv_u  = wcc_u + projN;
  float* wcc_ip = wvv_u + projN;
  float* wcc_in = wcc_ip + projN;
  float* o_u    = wcc_in + projN;
  float* o_ip   = o_u + oN;
  float* o_in   = o_ip + oN;
  float* u_fin  = o_in + oN;               // total ~5.3 MB of ws

  // contexts
  CVRS_user_ctx<<<B, 64, 0, stream>>>(ucv, ucl, E, Wc, Wv, cce_u, vceB);
  CVRS_item_ctx<<<B, 64, 0, stream>>>(items, icl, E, cce_ip);
  CVRS_item_ctx<<<B, 64, 0, stream>>>(nitem, nicl, E, cce_in);

  // WMMA projections: wcc = cce @ W^T, wvv = vce @ W_ht^T
  CVRS_proj<<<B / 16, 32, 0, stream>>>(cce_u, W, wcc_u);
  CVRS_proj<<<B / 16, 32, 0, stream>>>(vceB, Wht, wvv_u);
  CVRS_proj<<<B / 16, 32, 0, stream>>>(cce_ip, W, wcc_ip);
  CVRS_proj<<<B / 16, 32, 0, stream>>>(cce_in, W, wcc_in);

  // gather + attention + weighted sum
  const int tiles = B * Lct * Cct;         // 6144 waves per pass
  CVRS_layer_emb<<<tiles, 32, 0, stream>>>(uts, E, R, wcc_u, wvv_u, o_u, 1);
  CVRS_layer_emb<<<tiles, 32, 0, stream>>>(its, E, R, wcc_ip, nullptr, o_ip, 0);
  CVRS_layer_emb<<<tiles, 32, 0, stream>>>(nts, E, R, wcc_in, nullptr, o_in, 0);

  // finalize
  CVRS_user_final<<<B, 64, 0, stream>>>(o_u, u_fin);
  CVRS_item_final<<<B, 64, 0, stream>>>(o_ip, items, E, u_fin, out);
  CVRS_item_final<<<B, 64, 0, stream>>>(o_in, nitem, E, u_fin, out + B);
}
